// Conv2D_55989193670945
// MI455X (gfx1250) — matrix operand order repaired and checked by reading, not compiled, not run on hardware
//
#include <hip/hip_runtime.h>

typedef _Float16 f16;
typedef __attribute__((ext_vector_type(2)))  _Float16 v2h;
typedef __attribute__((ext_vector_type(8)))  _Float16 v8h;
typedef __attribute__((ext_vector_type(16))) _Float16 v16h;
typedef __attribute__((ext_vector_type(8)))  float    v8f;

#define IMG_H 1024
#define IMG_W 1024
#define BT_ROWS 64        // workgroup output tile rows
#define BT_COLS 64        // workgroup output tile cols
#define LDS_H   70        // 64 + 6 halo
#define LDS_W   80        // 64 + 16: A reads full K=0..31 span; row stride 160B (16B aligned)
#define NTHREADS 512      // 16 waves (wave32), 4x4 grid of 16x16 wave tiles
#define NPAIRS  (LDS_H * (LDS_W / 2))      // 2800 float pairs per tile
#define PPT     6                          // staging pairs per thread (6*512 >= 2800)
#define NB      8                          // batches per workgroup (amortizes setup)

__global__ __launch_bounds__(NTHREADS)
void conv7x7_wmma_f16(const float* __restrict__ X,
                      const float* __restrict__ W7,
                      float* __restrict__ Out)
{
    __shared__ f16   tile[2][LDS_H * LDS_W];
    __shared__ float wsh[49];

    const int tid  = threadIdx.x;
    const int lane = tid & 31;
    const int wave = tid >> 5;

    const int jb = blockIdx.x * BT_COLS;
    const int ib = blockIdx.y * BT_ROWS;
    const int b0 = blockIdx.z * NB;

    // uniform: whole halo tile inside the image?
    const bool interior = (ib >= 3) && (ib + (LDS_H - 3) <= IMG_H) &&
                          (jb >= 3) && (jb + (LDS_W - 3) <= IMG_W);

    // ---- staging: global -> regs (issued early to hide latency) ----
    float2 regs[PPT];
    auto load_pairs = [&](int bz, float2* rg) {
        const float* __restrict__ Xb = X + (size_t)bz * IMG_H * IMG_W;
        #pragma unroll
        for (int k = 0; k < PPT; ++k) {
            int idx = tid + k * NTHREADS;
            float v0 = 0.0f, v1 = 0.0f;
            if (idx < NPAIRS) {
                int r  = idx / (LDS_W / 2);
                int t  = idx - r * (LDS_W / 2);
                int gr = ib - 3 + r;
                int gc = jb - 3 + 2 * t;
                if (interior) {
                    const float* row = Xb + (size_t)gr * IMG_W + gc;
                    v0 = row[0];
                    v1 = row[1];
                } else if ((unsigned)gr < (unsigned)IMG_H) {
                    const float* row = Xb + (size_t)gr * IMG_W;
                    if ((unsigned)gc       < (unsigned)IMG_W) v0 = row[gc];
                    if ((unsigned)(gc + 1) < (unsigned)IMG_W) v1 = row[gc + 1];
                }
            }
            rg[k] = make_float2(v0, v1);
        }
    };

    // ---- staging: regs -> LDS (f16 convert, packed 4B stores) ----
    auto store_pairs = [&](f16* buf, const float2* rg) {
        #pragma unroll
        for (int k = 0; k < PPT; ++k) {
            int idx = tid + k * NTHREADS;
            if (idx < NPAIRS) {
                int r = idx / (LDS_W / 2);
                int t = idx - r * (LDS_W / 2);
                v2h p;
                p.x = (f16)rg[k].x;
                p.y = (f16)rg[k].y;
                *(v2h*)&buf[r * LDS_W + 2 * t] = p;
            }
        }
    };

    // ---- stage filter + first batch ----
    if (tid < 49) wsh[tid] = W7[tid];
    load_pairs(b0, regs);
    __syncthreads();                       // wsh visible for B build

    // ---- per-wave tile geometry ----
    const int wrow = (wave >> 2) * 16;     // 0,16,32,48
    const int wcol = (wave & 3)  * 16;     // 0,16,32,48
    const int m    = lane & 15;            // A row M / B col N / D col N
    const int kbB  = (lane >> 4) << 3;     // B: lanes 0-15 K=0-7 & 16-23, lanes 16-31 K=8-15 & 24-31
    const int koff = (lane >> 4) << 3;     // A: lanes 0-15 K=0-7 & 16-23, lanes 16-31 K=8-15 & 24-31

    // ---- banded-Toeplitz B fragments, built ONCE per wg (branch-free clamp+select) ----
    v16h Bf[7];
    #pragma unroll
    for (int u = 0; u < 7; ++u) {
        v16h bb;
        #pragma unroll
        for (int e = 0; e < 16; ++e) {
            int d  = kbB + e + ((e >> 3) << 3) - m;    // filter tap index
            int dc = d < 0 ? 0 : (d > 6 ? 6 : d);      // clamp -> unconditional load
            float wv = wsh[u * 7 + dc];
            bb[e] = (d == dc) ? (f16)wv : (f16)0.0f;
        }
        Bf[u] = bb;
    }

    store_pairs(tile[0], regs);
    __syncthreads();

    // ---- compute one 16x16 tile from LDS buffer, store f32 results ----
    auto compute_and_store = [&](const f16* buf, int bz) {
        v8f acc0 = {};
        v8f acc1 = {};
        #pragma unroll
        for (int u = 0; u < 7; ++u) {
            const int arow = wrow + m + u;                       // 0..69
            const f16* p = &buf[arow * LDS_W + wcol + koff];     // 16B aligned
            v8h r0 = *(const v8h*)p;                             // K = koff .. koff+7
            v8h r1 = *(const v8h*)(p + 16);                      // K = 16+koff .. +7
            v16h a = __builtin_shufflevector(r0, r1,
                     0, 1, 2, 3, 4, 5, 6, 7, 8, 9, 10, 11, 12, 13, 14, 15);
            if (u & 1)
                acc1 = __builtin_amdgcn_wmma_f32_16x16x32_f16(
                        false, a, false, Bf[u], (short)0, acc1, false, false);
            else
                acc0 = __builtin_amdgcn_wmma_f32_16x16x32_f16(
                        false, a, false, Bf[u], (short)0, acc0, false, false);
        }
        // D: VGPR r -> rows r / r+8 ; N = lane&15
        float* __restrict__ Ob = Out + (size_t)bz * IMG_H * IMG_W;
        const int orow0 = ib + wrow + ((lane < 16) ? 0 : 8);
        const int ocol  = jb + wcol + m;
        #pragma unroll
        for (int r = 0; r < 8; ++r) {
            float v = acc0[r] + acc1[r];
            Ob[(size_t)(orow0 + r) * IMG_W + ocol] = v;
        }
    };

    // ---- double-buffered batch loop ----
    for (int z = 0; z < NB - 1; ++z) {
        load_pairs(b0 + z + 1, regs);            // global loads in flight during compute
        compute_and_store(tile[z & 1], b0 + z);
        store_pairs(tile[(z + 1) & 1], regs);    // waits loads, fills other buffer
        __syncthreads();
    }
    compute_and_store(tile[(NB - 1) & 1], b0 + NB - 1);
}

extern "C" void kernel_launch(void* const* d_in, const int* in_sizes, int n_in,
                              void* d_out, int out_size, void* d_ws, size_t ws_size,
                              hipStream_t stream) {
    const float* X  = (const float*)d_in[0];   // [64, 1024, 1024] f32
    const float* W7 = (const float*)d_in[1];   // [7, 7] f32
    float* Out      = (float*)d_out;           // [64, 1024, 1024] f32

    dim3 grid(IMG_W / BT_COLS, IMG_H / BT_ROWS, 64 / NB);
    dim3 block(NTHREADS, 1, 1);
    conv7x7_wmma_f16<<<grid, block, 0, stream>>>(X, W7, Out);
}
